// PointNetSetAbstraction_3427383902478
// MI455X (gfx1250) — matrix operand order repaired and checked by reading, not compiled, not run on hardware
//
#include <hip/hip_runtime.h>

// ---------------------------------------------------------------------------
// PointNet++ Set Abstraction for MI455X (gfx1250, wave32, WMMA).
// B=8, N=8192, S=1024, K=32, MLP 6->64->64->128, training-mode BN.
//
// All matrix operands live in memory in *WMMA fragment layout* so the hot
// kernel is a pure b128 stream + v_wmma_f32_16x16x32_f16:
//   A-frags (weights, f16):  Afrag[mt][ck][lane][16]      (prepped once)
//   Activations (f16):       Act[ptile][mt][col][j]       j = channel & 15
//     - producer D-tile: lane writes 8 contiguous halves (one b128 store)
//     - consumer B-tile: lane reads 16 contiguous halves  (two b128 loads)
// Each wave processes TILES=8 consecutive point-tiles: weights/bias/BN params
// stay in registers, batch stats accumulate in registers and are reduced
// (bpermute + LDS atomics + 1 global atomic/channel/block) once per wave.
// BN+ReLU of layer l fused into layer l+1's B-fragment load; BN2+ReLU+max
// fused into the pooling kernel.
// ---------------------------------------------------------------------------

typedef __attribute__((ext_vector_type(16))) _Float16 v16h;
typedef __attribute__((ext_vector_type(8)))  _Float16 v8h;
typedef __attribute__((ext_vector_type(8)))  float    v8f;
typedef __attribute__((ext_vector_type(4)))  float    v4f;

constexpr int BB = 8;
constexpr int NN = 8192;
constexpr int SS = 1024;
constexpr int KK = 32;
constexpr int PP = BB * SS * KK;        // 262144 points through the MLP
constexpr int PTILES = PP / 16;         // 16384
constexpr float RADIUS2 = 0.4f * 0.4f;
constexpr float EPSBN   = 1e-5f;

// ---------------------------------------------------------------------------
// 1) Farthest point sampling: one block per batch, points held in registers.
// ---------------------------------------------------------------------------
__global__ __launch_bounds__(1024) void fps_kernel(
    const float* __restrict__ xyz, const unsigned char* __restrict__ mask,
    int* __restrict__ fidx)
{
  const int b = blockIdx.x;
  const int t = threadIdx.x;
  const int lane = t & 31, wv = t >> 5;
  const float* xb = xyz + (size_t)b * 3 * NN;
  const unsigned char* mb = mask + (size_t)b * NN;

  float xm[8], ym[8], zm[8], mind[8];
#pragma unroll
  for (int i = 0; i < 8; ++i) {
    const int n = t + i * 1024;
    const bool mv = mb[n] != 0;
    xm[i] = mv ? xb[n]          : 1e6f;
    ym[i] = mv ? xb[NN + n]     : 1e6f;
    zm[i] = mv ? xb[2 * NN + n] : 1e6f;
    mind[i] = 1e10f;
  }

  __shared__ float sval[32];
  __shared__ int   sidx[32];
  __shared__ float sc[3];
  __shared__ int   sfar;

  int far = 0;
  for (int s = 0; s < SS; ++s) {
    if (t == 0) fidx[b * SS + s] = far;
    if (t == (far & 1023)) {                 // owner broadcasts centroid
      const int i = far >> 10;
      sc[0] = xm[i]; sc[1] = ym[i]; sc[2] = zm[i];
    }
    __syncthreads();
    const float cx = sc[0], cy = sc[1], cz = sc[2];

    float bv = -1.0f; int bi = 0;
#pragma unroll
    for (int i = 0; i < 8; ++i) {
      const float dx = xm[i] - cx, dy = ym[i] - cy, dz = zm[i] - cz;
      const float d = dx * dx + dy * dy + dz * dz;
      const float m = fminf(mind[i], d);
      mind[i] = m;
      if (m > bv) { bv = m; bi = t + i * 1024; }  // ascending -> first max
    }
#pragma unroll
    for (int off = 16; off; off >>= 1) {
      const float ov = __shfl_xor(bv, off, 32);
      const int   oi = __shfl_xor(bi, off, 32);
      if (ov > bv || (ov == bv && oi < bi)) { bv = ov; bi = oi; }
    }
    if (lane == 0) { sval[wv] = bv; sidx[wv] = bi; }
    __syncthreads();
    if (wv == 0) {
      bv = sval[lane]; bi = sidx[lane];
#pragma unroll
      for (int off = 16; off; off >>= 1) {
        const float ov = __shfl_xor(bv, off, 32);
        const int   oi = __shfl_xor(bi, off, 32);
        if (ov > bv || (ov == bv && oi < bi)) { bv = ov; bi = oi; }
      }
      if (lane == 0) sfar = bi;
    }
    __syncthreads();
    far = sfar;
  }
}

// ---------------------------------------------------------------------------
// 2) new_xyz + new_mask outputs (tiny gather).
// ---------------------------------------------------------------------------
__global__ void newxyz_mask_kernel(
    const float* __restrict__ xyz, const unsigned char* __restrict__ mask,
    const int* __restrict__ fidx, float* __restrict__ out)
{
  const int id = blockIdx.x * blockDim.x + threadIdx.x;   // b*S+s
  if (id >= BB * SS) return;
  const int b = id >> 10, s = id & 1023;
  const int f = fidx[id];
#pragma unroll
  for (int c = 0; c < 3; ++c)
    out[((size_t)b * 3 + c) * SS + s] = xyz[((size_t)b * 3 + c) * NN + f];
  float* maskout = out + (size_t)BB * 3 * SS + (size_t)BB * 128 * SS;
  maskout[id] = mask[(size_t)b * NN + f] ? 1.0f : 0.0f;
}

// ---------------------------------------------------------------------------
// 3) Weight -> A-fragment prep (runs once, tiny). One wave per (mt, ck).
//    A (16x32 f16): row = mt*16+(lane&15); k = j + (lane>=16 ? 16 : 0)
// ---------------------------------------------------------------------------
__global__ __launch_bounds__(32) void wfrag_kernel(
    const float* __restrict__ W, _Float16* __restrict__ Afrag,
    int Cin, int Cout)
{
  const int nck = (Cin + 31) >> 5;
  const int wid = blockIdx.x;                 // = mt*nck + ck
  const int mt = wid / nck, ck = wid % nck;
  const int lane = threadIdx.x;
  const int col = lane & 15, hi = lane >> 4;
  v16h a;
#pragma unroll
  for (int j = 0; j < 16; ++j) {
    const int c = ck * 32 + j + hi * 16;
    a[j] = (c < Cin) ? (_Float16)W[(mt * 16 + col) * Cin + c] : (_Float16)0.f;
  }
  *(v16h*)(Afrag + ((size_t)wid * 32 + lane) * 16) = a;
}

// ---------------------------------------------------------------------------
// 4) Ball query (first-K ascending via ballot + prefix popcount) + gather
//    written directly in B-fragment activation layout (nmt_in = 2, channels
//    6..31 zero). One wave32 per centroid (= 32 points = 2 ptiles).
// ---------------------------------------------------------------------------
__global__ __launch_bounds__(256) void ballq_gather_kernel(
    const float* __restrict__ xyz, const float* __restrict__ pts,
    const int* __restrict__ fidx, _Float16* __restrict__ x0)
{
  __shared__ int gbuf[8][KK];
  const int wid  = (blockIdx.x * 256 + threadIdx.x) >> 5;  // = b*S + s
  const int lane = threadIdx.x & 31;
  const int wv   = threadIdx.x >> 5;
  const int b = wid >> 10;
  const float* xb = xyz + (size_t)b * 3 * NN;
  const float* pb = pts + (size_t)b * 3 * NN;

  const int f = fidx[wid];
  const float cx = xb[f], cy = xb[NN + f], cz = xb[2 * NN + f];

  int found = 0;
  for (int n0 = 0; n0 < NN && found < KK; n0 += 32) {
    const int n = n0 + lane;
    const float dx = xb[n] - cx, dy = xb[NN + n] - cy, dz = xb[2 * NN + n] - cz;
    const bool inball = (dx * dx + dy * dy + dz * dz) <= RADIUS2;
    const unsigned mb = __builtin_amdgcn_ballot_w32(inball);
    const int pos = found + __popc(mb & ((1u << lane) - 1u));
    if (inball && pos < KK) gbuf[wv][pos] = n;
    found += __popc(mb);
  }
  asm volatile("s_wait_dscnt 0" ::: "memory");   // wave-local LDS ordering
  if (found > KK) found = KK;                    // found >= 1 (center in-ball)
  const int g = (lane < found) ? gbuf[wv][lane] : gbuf[wv][0];

  // point p = wid*32 + lane -> ptile = wid*2 + hi, col = lane&15
  const int ptile = wid * 2 + (lane >> 4);
  const int col   = lane & 15;
  v16h vals;
#pragma unroll
  for (int j = 0; j < 16; ++j) vals[j] = (_Float16)0.f;
  vals[0] = (_Float16)(xb[g] - cx);
  vals[1] = (_Float16)(xb[NN + g] - cy);
  vals[2] = (_Float16)(xb[2 * NN + g] - cz);
  vals[3] = (_Float16)pb[g];
  vals[4] = (_Float16)pb[NN + g];
  vals[5] = (_Float16)pb[2 * NN + g];
  v16h zero;
#pragma unroll
  for (int j = 0; j < 16; ++j) zero[j] = (_Float16)0.f;
  *(v16h*)(x0 + (((size_t)ptile * 2 + 0) * 16 + col) * 16) = vals;   // mt 0
  *(v16h*)(x0 + (((size_t)ptile * 2 + 1) * 16 + col) * 16) = zero;   // mt 1
}

// ---------------------------------------------------------------------------
// 5) WMMA MLP layer. One wave -> TILES consecutive 16(out-ch) x 16(point)
//    f32 tiles with weights/bias/BN params register-resident; batch stats
//    accumulate in registers, reduced once per wave at the end.
// ---------------------------------------------------------------------------
template <int NCHUNKS, bool BN, int TILES>
__global__ __launch_bounds__(256) void mlp_wmma_kernel(
    const _Float16* __restrict__ Xfrag,   // [ptile][2*NCHUNKS][col][16]
    const _Float16* __restrict__ Afrag,   // [mt][NCHUNKS][lane][16]
    const float* __restrict__ bias,
    const float* __restrict__ scaleIn, const float* __restrict__ shiftIn,
    _Float16* __restrict__ Yfrag,         // [ptile][nmt_out][col][16]
    float* __restrict__ gsum, float* __restrict__ gsumsq, int nmt_out)
{
  constexpr int NMT_IN = NCHUNKS * 2;
  __shared__ float lsum[16], lsq[16];
  const int t = threadIdx.x;
  const int lane = t & 31, wv = t >> 5;
  if (t < 16) { lsum[t] = 0.f; lsq[t] = 0.f; }
  __syncthreads();

  const int mt = blockIdx.y;
  const int m0 = mt * 16;
  const int ptile0 = (blockIdx.x * 8 + wv) * TILES;
  const int col = lane & 15, hi = lane >> 4;

  // Wave-resident operands ------------------------------------------------
  v16h a[NCHUNKS];
#pragma unroll
  for (int ck = 0; ck < NCHUNKS; ++ck)
    a[ck] = *(const v16h*)(Afrag + ((size_t)(mt * NCHUNKS + ck) * 32 + lane) * 16);

  v8f bias8;
  {
    const v4f* b4 = (const v4f*)(bias + m0 + hi * 8);
    const v4f blo = b4[0], bhi = b4[1];
#pragma unroll
    for (int v = 0; v < 4; ++v) { bias8[v] = blo[v]; bias8[v + 4] = bhi[v]; }
  }

  v4f scr[NCHUNKS][4], shr[NCHUNKS][4];
  if constexpr (BN) {
#pragma unroll
    for (int ck = 0; ck < NCHUNKS; ++ck) {
      const int cbase = ck * 32 + hi * 16;
      const v4f* s4 = (const v4f*)(scaleIn + cbase);
      const v4f* h4 = (const v4f*)(shiftIn + cbase);
#pragma unroll
      for (int q = 0; q < 4; ++q) { scr[ck][q] = s4[q]; shr[ck][q] = h4[q]; }
    }
  }

  float s1[8], s2[8];
#pragma unroll
  for (int v = 0; v < 8; ++v) { s1[v] = 0.f; s2[v] = 0.f; }

  // Tile loop --------------------------------------------------------------
#pragma unroll
  for (int tt = 0; tt < TILES; ++tt) {
    const size_t ptile = ptile0 + tt;
    if (tt + 1 < TILES) {   // prefetch next tile's B-fragment cacheline
      __builtin_prefetch(
          (const void*)(Xfrag + (((ptile + 1) * NMT_IN + hi) * 16 + col) * 16),
          0, 1);
    }
    v8f acc;
#pragma unroll
    for (int v = 0; v < 8; ++v) acc[v] = bias8[v];

#pragma unroll
    for (int ck = 0; ck < NCHUNKS; ++ck) {
      const v16h x = *(const v16h*)(
          Xfrag + ((ptile * NMT_IN + ck * 2 + hi) * 16 + col) * 16);
      v16h bfrag;
      if constexpr (BN) {
#pragma unroll
        for (int q = 0; q < 4; ++q) {
          const v4f sc = scr[ck][q], sh = shr[ck][q];
#pragma unroll
          for (int r = 0; r < 4; ++r) {
            const float xv = (float)x[q * 4 + r];
            bfrag[q * 4 + r] = (_Float16)fmaxf(xv * sc[r] + sh[r], 0.f);
          }
        }
      } else {
        bfrag = x;                          // layer 0: raw input, no BN
      }
      acc = __builtin_amdgcn_wmma_f32_16x16x32_f16(
          false, a[ck], false, bfrag, (short)0, acc, false, false);
    }

    // D-tile -> fragment-layout store: 8 contiguous halves per lane.
    v8h outv;
#pragma unroll
    for (int v = 0; v < 8; ++v) outv[v] = (_Float16)acc[v];
    *(v8h*)(Yfrag + ((ptile * nmt_out + mt) * 16 + col) * 16 + hi * 8) = outv;

    // register-resident stats accumulation (reduced once per wave below)
#pragma unroll
    for (int v = 0; v < 8; ++v) {
      const float y = acc[v];
      s1[v] += y;
      s2[v] += y * y;
    }
  }

  // Per-wave stats reduction: 16 cols -> 1, then LDS + global atomics.
#pragma unroll
  for (int v = 0; v < 8; ++v) {
    float a1 = s1[v], a2 = s2[v];
#pragma unroll
    for (int off = 1; off < 16; off <<= 1) {
      a1 += __shfl_xor(a1, off, 32);
      a2 += __shfl_xor(a2, off, 32);
    }
    if (col == 0) {
      atomicAdd(&lsum[v + hi * 8], a1);
      atomicAdd(&lsq [v + hi * 8], a2);
    }
  }
  __syncthreads();
  if (t < 16) {
    atomicAdd(&gsum  [m0 + t], lsum[t]);
    atomicAdd(&gsumsq[m0 + t], lsq[t]);
  }
}

// ---------------------------------------------------------------------------
// 6) BN finalize: scale = g/sqrt(var+eps), shift = beta - mean*scale.
// ---------------------------------------------------------------------------
__global__ void bn_finalize_kernel(
    const float* __restrict__ gsum, const float* __restrict__ gsumsq,
    const float* __restrict__ g, const float* __restrict__ be,
    float* __restrict__ scale, float* __restrict__ shift, int Cout)
{
  const int o = threadIdx.x;
  if (o >= Cout) return;
  const float inv_n = 1.0f / (float)PP;
  const float m = gsum[o] * inv_n;
  const float v = fmaxf(gsumsq[o] * inv_n - m * m, 0.f);
  const float inv = rsqrtf(v + EPSBN);
  const float sc = g[o] * inv;
  scale[o] = sc;
  shift[o] = be[o] - m * sc;
}

// ---------------------------------------------------------------------------
// 7) BN2+ReLU + max over K=32: one wave per (bs, mt); b128 fragment loads,
//    cross-lane max over the 32 points (16 cols x 2 ptiles).
// ---------------------------------------------------------------------------
__global__ __launch_bounds__(256) void maxpool_kernel(
    const _Float16* __restrict__ Yfrag, const float* __restrict__ scale,
    const float* __restrict__ shift, float* __restrict__ outpts)
{
  const int wid  = blockIdx.x * 8 + (threadIdx.x >> 5);
  const int lane = threadIdx.x & 31;
  const int col = lane & 15, hi = lane >> 4;
  const int bs = wid & (BB * SS - 1);
  const int mt = wid >> 13;                      // 0..7
  const int ptile = bs * 2 + hi;
  const v16h x = *(const v16h*)(Yfrag + (((size_t)ptile * 8 + mt) * 16 + col) * 16);
  const v4f* s4 = (const v4f*)(scale + mt * 16);
  const v4f* h4 = (const v4f*)(shift + mt * 16);
  const int b = bs >> 10, s = bs & 1023;
#pragma unroll
  for (int q = 0; q < 4; ++q) {
    const v4f sc = s4[q], sh = h4[q];
#pragma unroll
    for (int r = 0; r < 4; ++r) {
      const int j = q * 4 + r;
      float v = fmaxf((float)x[j] * sc[r] + sh[r], 0.f);
#pragma unroll
      for (int off = 16; off; off >>= 1) v = fmaxf(v, __shfl_xor(v, off, 32));
      if (lane == 0)
        outpts[((size_t)b * 128 + mt * 16 + j) * SS + s] = v;
    }
  }
}

__global__ void zero_stats_kernel(float* stats) {
  const int t = blockIdx.x * blockDim.x + threadIdx.x;
  if (t < 3 * 512) stats[t] = 0.f;
}

// ---------------------------------------------------------------------------
extern "C" void kernel_launch(void* const* d_in, const int* in_sizes, int n_in,
                              void* d_out, int out_size, void* d_ws, size_t ws_size,
                              hipStream_t stream)
{
  const float* xyz = (const float*)d_in[0];
  const float* pts = (const float*)d_in[1];
  const unsigned char* mask = (const unsigned char*)d_in[2];
  const float* W0 = (const float*)d_in[3];  const float* b0 = (const float*)d_in[4];
  const float* g0 = (const float*)d_in[5];  const float* be0 = (const float*)d_in[6];
  const float* W1 = (const float*)d_in[7];  const float* b1 = (const float*)d_in[8];
  const float* g1 = (const float*)d_in[9];  const float* be1 = (const float*)d_in[10];
  const float* W2 = (const float*)d_in[11]; const float* b2 = (const float*)d_in[12];
  const float* g2 = (const float*)d_in[13]; const float* be2 = (const float*)d_in[14];

  // Workspace layout (~145 MB total):
  char* ws = (char*)d_ws;
  int*      fidx  = (int*)ws;                                       // 32 KB
  float*    stats = (float*)(ws + (size_t)32768);                   // 6 KB
  _Float16* Af0 = (_Float16*)(ws + (size_t)40960);                  // 4 KB
  _Float16* Af1 = (_Float16*)(ws + (size_t)49152);                  // 8 KB
  _Float16* Af2 = (_Float16*)(ws + (size_t)65536);                  // 16 KB
  _Float16* x0 = (_Float16*)(ws + (size_t)131072);                  // PT*2*512B = 16 MB
  _Float16* y0 = (_Float16*)((char*)x0 + (size_t)PTILES * 2 * 512); // PT*4*512B = 32 MB
  _Float16* y1 = (_Float16*)((char*)y0 + (size_t)PTILES * 4 * 512); // 32 MB
  _Float16* y2 = (_Float16*)((char*)y1 + (size_t)PTILES * 4 * 512); // PT*8*512B = 64 MB

  float* out    = (float*)d_out;
  float* outpts = out + (size_t)BB * 3 * SS;

  zero_stats_kernel<<<6, 256, 0, stream>>>(stats);
  wfrag_kernel<<<4,  32, 0, stream>>>(W0, Af0, 6,  64);   // 4 mt x 1 ck
  wfrag_kernel<<<8,  32, 0, stream>>>(W1, Af1, 64, 64);   // 4 mt x 2 ck
  wfrag_kernel<<<16, 32, 0, stream>>>(W2, Af2, 64, 128);  // 8 mt x 2 ck

  fps_kernel<<<BB, 1024, 0, stream>>>(xyz, mask, fidx);
  newxyz_mask_kernel<<<(BB * SS + 255) / 256, 256, 0, stream>>>(xyz, mask, fidx, out);
  ballq_gather_kernel<<<(BB * SS) / 8, 256, 0, stream>>>(xyz, pts, fidx, x0);

  constexpr int TILES = 8;                       // point-tiles per wave
  const dim3 blk(256);
  const int gx = PTILES / (8 * TILES);           // 256 blocks in x
  // Layer 0: 6 -> 64 (raw input, no BN on load)
  mlp_wmma_kernel<1, false, TILES><<<dim3(gx, 4), blk, 0, stream>>>(
      x0, Af0, b0, nullptr, nullptr, y0, stats + 0, stats + 128, 4);
  bn_finalize_kernel<<<1, 128, 0, stream>>>(stats + 0, stats + 128, g0, be0,
                                            stats + 256, stats + 384, 64);
  // Layer 1: 64 -> 64 (fused BN0+ReLU on load)
  mlp_wmma_kernel<2, true, TILES><<<dim3(gx, 4), blk, 0, stream>>>(
      y0, Af1, b1, stats + 256, stats + 384, y1, stats + 512, stats + 640, 4);
  bn_finalize_kernel<<<1, 128, 0, stream>>>(stats + 512, stats + 640, g1, be1,
                                            stats + 768, stats + 896, 64);
  // Layer 2: 64 -> 128 (fused BN1+ReLU on load)
  mlp_wmma_kernel<2, true, TILES><<<dim3(gx, 8), blk, 0, stream>>>(
      y1, Af2, b2, stats + 768, stats + 896, y2, stats + 1024, stats + 1152, 8);
  bn_finalize_kernel<<<1, 128, 0, stream>>>(stats + 1024, stats + 1152, g2, be2,
                                            stats + 1280, stats + 1408, 128);
  // BN2+ReLU + max over K
  maxpool_kernel<<<(8 * BB * SS) / 8, 256, 0, stream>>>(
      y2, stats + 1280, stats + 1408, outpts);
}
